// Discriminator_28183575396353
// MI455X (gfx1250) — matrix operand order repaired and checked by reading, not compiled, not run on hardware
//
#include <hip/hip_runtime.h>

typedef __attribute__((ext_vector_type(16))) __bf16 v16bf;
typedef __attribute__((ext_vector_type(8)))  __bf16 v8bf;
typedef __attribute__((ext_vector_type(8)))  float  v8f;

#define B_SZ 16
#define N_SZ 1024
#define KNN_K 20
#define NEG_INF (-3.0e38f)

__device__ __forceinline__ float lrelu(float v) { return v >= 0.f ? v : 0.2f * v; }

// CAS-based float max; works on LDS and global pointers.
__device__ __forceinline__ void atomicMaxF(float* addr, float val) {
    unsigned int* ua = (unsigned int*)addr;
    unsigned int old = *ua;
    while (val > __uint_as_float(old)) {
        unsigned int assumed = old;
        old = atomicCAS(ua, assumed, __float_as_uint(val));
        if (old == assumed) break;
    }
}

// Concatenate two 8-element bf16 vectors (two 128-bit loads) into a fragment.
__device__ __forceinline__ v16bf cat16(v8bf lo, v8bf hi) {
    return __builtin_shufflevector(lo, hi, 0, 1, 2, 3, 4, 5, 6, 7,
                                          8, 9, 10, 11, 12, 13, 14, 15);
}

// ---------------------------------------------------------------------------
// Transpose x [B,N,6] -> xt [B,6,N]
// ---------------------------------------------------------------------------
__global__ void transpose_x_kernel(const float* __restrict__ x, float* __restrict__ xt) {
    int i = blockIdx.x * 256 + threadIdx.x;
    int total = B_SZ * 6 * N_SZ;
    if (i >= total) return;
    int b = i / (6 * N_SZ);
    int rem = i % (6 * N_SZ);
    int c = rem / N_SZ, n = rem % N_SZ;
    xt[i] = x[((size_t)b * N_SZ + n) * 6 + c];
}

// ---------------------------------------------------------------------------
// Per-point squared norms of feat [B,D,N] -> norms [B*N]
// ---------------------------------------------------------------------------
__global__ void norms_kernel(const float* __restrict__ x, float* __restrict__ norms, int D) {
    int i = blockIdx.x * 256 + threadIdx.x;
    if (i >= B_SZ * N_SZ) return;
    int b = i / N_SZ, n = i % N_SZ;
    float s = 0.f;
    for (int c = 0; c < D; ++c) {
        float v = x[((size_t)b * D + c) * N_SZ + n];
        s += v * v;
    }
    norms[i] = s;
}

// ---------------------------------------------------------------------------
// KNN: one thread per query point, register top-20 (score = 2*dot - ||m||^2).
// Candidate features tiled through LDS (32 candidates).
// ---------------------------------------------------------------------------
template <int D>
__global__ void __launch_bounds__(256, 1) knn_kernel(const float* __restrict__ x,
                                                     const float* __restrict__ norms,
                                                     int* __restrict__ idx) {
    constexpr int T = 32;
    extern __shared__ char smemraw[];
    float* sm    = (float*)smemraw;     // [D][T] candidate features, c-major
    float* snorm = sm + D * T;          // [T]
    int tid = threadIdx.x;
    int b = blockIdx.x / (N_SZ / 256);
    int n = (blockIdx.x % (N_SZ / 256)) * 256 + tid;
    const size_t xb = (size_t)b * D * N_SZ;

    float q[D];
#pragma unroll
    for (int c = 0; c < D; ++c) q[c] = x[xb + (size_t)c * N_SZ + n];

    float sc[KNN_K];
    int   si[KNN_K];
#pragma unroll
    for (int t = 0; t < KNN_K; ++t) { sc[t] = NEG_INF; si[t] = 0; }

    for (int m0 = 0; m0 < N_SZ; m0 += T) {
        __syncthreads();
        for (int i = tid; i < D * T; i += 256) {
            int c = i / T, mt = i % T;
            sm[i] = x[xb + (size_t)c * N_SZ + m0 + mt];
        }
        if (tid < T) snorm[tid] = norms[(size_t)b * N_SZ + m0 + tid];
        __syncthreads();
        for (int mt = 0; mt < T; ++mt) {
            float dot = 0.f;
#pragma unroll
            for (int c = 0; c < D; ++c) dot += q[c] * sm[c * T + mt];
            float score = 2.f * dot - snorm[mt];
            int m = m0 + mt;
            if (score > sc[KNN_K - 1]) {
#pragma unroll
                for (int t = KNN_K - 1; t >= 0; --t) {
                    if (score > sc[t]) {
                        bool shift = (t > 0) && (score > sc[t - 1]);
                        sc[t] = shift ? sc[t - 1] : score;
                        si[t] = shift ? si[t - 1] : m;
                    }
                }
            }
        }
    }
    for (int t = 0; t < KNN_K; ++t)
        idx[((size_t)b * N_SZ + n) * KNN_K + t] = si[t];
}

// ---------------------------------------------------------------------------
// EdgeConv: per block, 4 points x 32 padded neighbor rows. Edge features and
// bf16 weights staged in LDS (both K-contiguous), fragments via 128-bit LDS
// loads, GEMM via v_wmma_f32_16x16x32_bf16, leaky-relu, max over neighbors.
// x: [B,D,N] f32, W: [OUT,2D] f32, out: [B,OUT,N] f32.
// Also writes its bf16 slice of cat [B,N,512] at channel offset `coff`.
// ---------------------------------------------------------------------------
template <int D, int OUT>
__global__ void __launch_bounds__(256, 1)
edge_conv_kernel(const float* __restrict__ x, const int* __restrict__ idx,
                 const float* __restrict__ W, float* __restrict__ out,
                 __bf16* __restrict__ catb, int coff) {
    constexpr int KD  = ((2 * D + 31) / 32) * 32;   // padded K dimension
    constexpr int PPB = 4;                           // points per block
    extern __shared__ char smemraw[];
    __bf16* E    = (__bf16*)smemraw;                                    // [PPB*32][KD]
    __bf16* Wl   = (__bf16*)(smemraw + (size_t)PPB * 32 * KD * 2);      // [OUT][KD]
    float*  omax = (float*)((char*)Wl + (size_t)OUT * KD * 2);          // [PPB][OUT]
    int*    nidx = (int*)(omax + PPB * OUT);                            // [PPB*32]

    int tid = threadIdx.x;
    int b = blockIdx.x / (N_SZ / PPB);
    int pbase = (blockIdx.x % (N_SZ / PPB)) * PPB;

    for (int i = tid; i < PPB * 32; i += 256) {
        int p = i >> 5, s = i & 31;
        int ss = s < KNN_K ? s : (KNN_K - 1);
        nidx[i] = idx[((size_t)b * N_SZ + (pbase + p)) * KNN_K + ss];
    }
    for (int i = tid; i < PPB * OUT; i += 256) omax[i] = NEG_INF;
    // stage weights as bf16, K-contiguous per output column, zero-padded K
    for (int i = tid; i < OUT * KD; i += 256) {
        int o = i / KD, k = i % KD;
        Wl[i] = (__bf16)((k < 2 * D) ? W[(size_t)o * (2 * D) + k] : 0.f);
    }
    __syncthreads();

    const size_t xb = (size_t)b * D * N_SZ;
    for (int i = tid; i < PPB * 32 * KD; i += 256) {
        int row = i / KD, c = i % KD;
        int n = pbase + (row >> 5);
        float v = 0.f;
        if (c < D) {
            int j = nidx[row];
            v = x[xb + (size_t)c * N_SZ + j] - x[xb + (size_t)c * N_SZ + n];
        } else if (c < 2 * D) {
            v = x[xb + (size_t)(c - D) * N_SZ + n];
        }
        E[(size_t)row * KD + c] = (__bf16)v;
    }
    __syncthreads();

    int wave = tid >> 5, lane = tid & 31, r = lane & 15;
    int half8 = (lane & 16) ? 8 : 0;    // A-fragment K sub-offset
    int kh    = (lane & 16) ? 16 : 0;   // unused; B fragment also uses half8
    int rowBase = wave * 16;            // 8 waves cover PPB*32 = 128 rows

    // Hoist all A fragments for this wave's row tile (<= 8 frags = 64 VGPRs)
    v16bf afrag[KD / 32];
    const __bf16* erow = E + (size_t)(rowBase + r) * KD;
#pragma unroll
    for (int ks = 0; ks < KD / 32; ++ks) {
        int ka = ks * 32 + half8;
        afrag[ks] = cat16(*(const v8bf*)(erow + ka), *(const v8bf*)(erow + ka + 16));
    }

#pragma unroll 1
    for (int ct = 0; ct < OUT / 16; ++ct) {
        const __bf16* wcol = Wl + (size_t)(ct * 16 + r) * KD;
        v8f acc = {};
#pragma unroll
        for (int ks = 0; ks < KD / 32; ++ks) {
            const __bf16* wp = wcol + ks * 32 + half8;
            v16bf w = cat16(*(const v8bf*)wp, *(const v8bf*)(wp + 16));
            acc = __builtin_amdgcn_wmma_f32_16x16x32_bf16(false, afrag[ks], false, w,
                                                          (short)0, acc, false, false);
        }
        float m = NEG_INF;
#pragma unroll
        for (int v = 0; v < 8; ++v) m = fmaxf(m, lrelu(acc[v]));
        m = fmaxf(m, __shfl_xor(m, 16, 32));
        if (lane < 16) atomicMaxF(&omax[(rowBase >> 5) * OUT + ct * 16 + r], m);
    }
    __syncthreads();

    for (int i = tid; i < PPB * OUT; i += 256) {
        int p = i / OUT, o = i % OUT;
        float v = omax[i];
        int n = pbase + p;
        out[((size_t)b * OUT + o) * N_SZ + n] = v;
        catb[((size_t)b * N_SZ + n) * 512 + coff + o] = (__bf16)v;
    }
}

// ---------------------------------------------------------------------------
// h = lrelu(W4 @ cat) then g = max over points.  WMMA over K=512, bf16.
// Grid: B * (1024/16) blocks; each block owns one batch + 16 output cols and
// reduces the point-max locally in LDS (no global atomics).
// Register blocking: each B fragment reused across 4 row tiles; K walked in
// chunks of 128 (4 live W fragments) to stay within the register file.
// ---------------------------------------------------------------------------
__global__ void __launch_bounds__(256, 1)
gmax_kernel(const __bf16* __restrict__ cat, const float* __restrict__ W4,
            float* __restrict__ g) {
    extern __shared__ char smemraw[];
    __bf16* Wt     = (__bf16*)smemraw;                  // [16 cols][512] K-contiguous
    float*  colred = (float*)(smemraw + 16 * 512 * 2);  // [16]
    int tid = threadIdx.x;
    int b = blockIdx.x >> 6;
    int colBase = (blockIdx.x & 63) * 16;

    for (int i = tid; i < 16 * 512; i += 256) {
        int nn = i >> 9, k = i & 511;
        Wt[i] = (__bf16)W4[(size_t)(colBase + nn) * 512 + k];
    }
    if (tid < 16) colred[tid] = NEG_INF;
    __syncthreads();

    int wave = tid >> 5, lane = tid & 31, r = lane & 15;
    int half8 = (lane & 16) ? 8 : 0;
    int kh    = (lane & 16) ? 16 : 0;
    const __bf16* wcol  = Wt + (size_t)r * 512;
    const __bf16* abase = cat + (size_t)b * N_SZ * 512;
    float cmax = NEG_INF;

#pragma unroll 1
    for (int grp = 0; grp < 2; ++grp) {          // 2 groups of 4 row tiles
        v8f accs[4];
#pragma unroll
        for (int it = 0; it < 4; ++it) accs[it] = (v8f){};
#pragma unroll 1
        for (int kc = 0; kc < 4; ++kc) {         // K chunks of 128
            int kbase = kc * 128;
            v16bf wf[4];
#pragma unroll
            for (int ks = 0; ks < 4; ++ks) {
                const __bf16* wp = wcol + kbase + ks * 32 + half8;
                wf[ks] = cat16(*(const v8bf*)wp, *(const v8bf*)(wp + 16));
            }
#pragma unroll
            for (int it = 0; it < 4; ++it) {
                int m = (wave * 8 + grp * 4 + it) * 16 + r;
                const __bf16* arow = abase + (size_t)m * 512 + kbase;
#pragma unroll
                for (int ks = 0; ks < 4; ++ks) {
                    int ka = ks * 32 + half8;
                    v16bf a = cat16(*(const v8bf*)(arow + ka),
                                    *(const v8bf*)(arow + ka + 16));
                    accs[it] = __builtin_amdgcn_wmma_f32_16x16x32_bf16(
                        false, a, false, wf[ks], (short)0, accs[it], false, false);
                }
            }
        }
#pragma unroll
        for (int it = 0; it < 4; ++it)
#pragma unroll
            for (int v = 0; v < 8; ++v) cmax = fmaxf(cmax, lrelu(accs[it][v]));
    }
    cmax = fmaxf(cmax, __shfl_xor(cmax, 16, 32));
    if (lane < 16) atomicMaxF(&colred[r], cmax);
    __syncthreads();
    if (tid < 16) g[b * 1024 + colBase + tid] = colred[tid];
}

// ---------------------------------------------------------------------------
// Final MLP (B=16): tiny, one block, LDS staged.
// ---------------------------------------------------------------------------
__global__ void final_mlp_kernel(const float* __restrict__ g,  const float* __restrict__ y,
                                 const float* __restrict__ F0w, const float* __restrict__ F0b,
                                 const float* __restrict__ F1w, const float* __restrict__ F1b,
                                 const float* __restrict__ L0,  const float* __restrict__ L1,
                                 const float* __restrict__ L2w, const float* __restrict__ L2b,
                                 float* __restrict__ out) {
    extern __shared__ char smemraw[];
    float* z  = (float*)smemraw;     // [16][1088]
    float* z1 = z  + 16 * 1088;      // [16][512]
    float* z2 = z1 + 16 * 512;       // [16][256]
    float* t0 = z2 + 16 * 256;       // [16][16]
    int tid = threadIdx.x;

    for (int i = tid; i < 16 * 1024; i += 256) {
        int row = i >> 10, c = i & 1023;
        z[row * 1088 + c] = g[row * 1024 + c];
    }
    {   // ye stage 1: [16,16]
        int row = tid >> 4, col = tid & 15;
        float s = F0b[col];
        for (int c = 0; c < 16; ++c) s += y[row * 16 + c] * F0w[col * 16 + c];
        t0[row * 16 + col] = lrelu(s);
    }
    __syncthreads();
    for (int i = tid; i < 16 * 64; i += 256) {   // ye stage 2 -> z[:,1024:]
        int row = i >> 6, col = i & 63;
        float s = F1b[col];
        for (int c = 0; c < 16; ++c) s += t0[row * 16 + c] * F1w[col * 16 + c];
        z[row * 1088 + 1024 + col] = lrelu(s);
    }
    __syncthreads();
    for (int i = tid; i < 16 * 512; i += 256) {  // L0
        int row = i >> 9, col = i & 511;
        float s = 0.f;
        for (int c = 0; c < 1088; ++c) s += z[row * 1088 + c] * L0[(size_t)col * 1088 + c];
        z1[row * 512 + col] = lrelu(s);
    }
    __syncthreads();
    for (int i = tid; i < 16 * 256; i += 256) {  // L1
        int row = i >> 8, col = i & 255;
        float s = 0.f;
        for (int c = 0; c < 512; ++c) s += z1[row * 512 + c] * L1[(size_t)col * 512 + c];
        z2[row * 256 + col] = lrelu(s);
    }
    __syncthreads();
    if (tid < 16) {                              // L2
        float s = L2b[0];
        for (int c = 0; c < 256; ++c) s += z2[tid * 256 + c] * L2w[c];
        out[tid] = s;
    }
}

// ---------------------------------------------------------------------------
// Host launch
// ---------------------------------------------------------------------------
static constexpr size_t edge_lds(int D, int OUT) {
    int KD = ((2 * D + 31) / 32) * 32;
    return (size_t)4 * 32 * KD * 2      // E
         + (size_t)OUT * KD * 2         // Wl
         + (size_t)4 * OUT * 4          // omax
         + (size_t)4 * 32 * 4;          // nidx
}

extern "C" void kernel_launch(void* const* d_in, const int* in_sizes, int n_in,
                              void* d_out, int out_size, void* d_ws, size_t ws_size,
                              hipStream_t stream) {
    (void)in_sizes; (void)n_in; (void)out_size; (void)ws_size;
    const float* x   = (const float*)d_in[0];
    const float* y   = (const float*)d_in[1];
    const float* W0  = (const float*)d_in[2];
    const float* W1  = (const float*)d_in[3];
    const float* W2  = (const float*)d_in[4];
    const float* W3  = (const float*)d_in[5];
    const float* W4  = (const float*)d_in[6];
    const float* L0  = (const float*)d_in[7];
    const float* L1  = (const float*)d_in[8];
    const float* L2w = (const float*)d_in[9];
    const float* L2b = (const float*)d_in[10];
    const float* F0w = (const float*)d_in[11];
    const float* F0b = (const float*)d_in[12];
    const float* F1w = (const float*)d_in[13];
    const float* F1b = (const float*)d_in[14];
    float* outp = (float*)d_out;

    // Workspace carve (floats)
    float* xt    = (float*)d_ws;                        // 16*6*1024
    float* f0    = xt + 16 * 6 * 1024;                  // 16*64*1024
    float* f1    = f0 + 16 * 64 * 1024;                 // 16*64*1024
    float* f2    = f1 + 16 * 64 * 1024;                 // 16*128*1024
    float* f3    = f2 + 16 * 128 * 1024;                // 16*256*1024
    float* g     = f3 + 16 * 256 * 1024;                // 16*1024
    float* norms = g + 16 * 1024;                       // 16*1024
    int*   idx   = (int*)(norms + 16 * 1024);           // 16*1024*20
    __bf16* catb = (__bf16*)(idx + 16 * 1024 * KNN_K);  // 16*1024*512 (point-major)

    transpose_x_kernel<<<(B_SZ * 6 * N_SZ + 255) / 256, 256, 0, stream>>>(x, xt);

    // Layer 0: d=6 -> 64  (cat channels [0,64))
    norms_kernel<<<64, 256, 0, stream>>>(xt, norms, 6);
    knn_kernel<6><<<64, 256, (6 * 32 + 32) * sizeof(float), stream>>>(xt, norms, idx);
    edge_conv_kernel<6, 64><<<B_SZ * N_SZ / 4, 256, edge_lds(6, 64), stream>>>(
        xt, idx, W0, f0, catb, 0);

    // Layer 1: d=64 -> 64  (cat channels [64,128))
    norms_kernel<<<64, 256, 0, stream>>>(f0, norms, 64);
    knn_kernel<64><<<64, 256, (64 * 32 + 32) * sizeof(float), stream>>>(f0, norms, idx);
    edge_conv_kernel<64, 64><<<B_SZ * N_SZ / 4, 256, edge_lds(64, 64), stream>>>(
        f0, idx, W1, f1, catb, 64);

    // Layer 2: d=64 -> 128  (cat channels [128,256))
    norms_kernel<<<64, 256, 0, stream>>>(f1, norms, 64);
    knn_kernel<64><<<64, 256, (64 * 32 + 32) * sizeof(float), stream>>>(f1, norms, idx);
    edge_conv_kernel<64, 128><<<B_SZ * N_SZ / 4, 256, edge_lds(64, 128), stream>>>(
        f1, idx, W2, f2, catb, 128);

    // Layer 3: d=128 -> 256  (cat channels [256,512))
    norms_kernel<<<64, 256, 0, stream>>>(f2, norms, 128);
    knn_kernel<128><<<64, 256, (128 * 32 + 32) * sizeof(float), stream>>>(f2, norms, idx);
    edge_conv_kernel<128, 256><<<B_SZ * N_SZ / 4, 256, edge_lds(128, 256), stream>>>(
        f2, idx, W3, f3, catb, 256);

    // W4 projection + global max over points
    gmax_kernel<<<B_SZ * 64, 256, 16 * 512 * 2 + 16 * sizeof(float), stream>>>(catb, W4, g);

    // Final MLP
    size_t flds = (size_t)(16 * 1088 + 16 * 512 + 16 * 256 + 16 * 16) * sizeof(float);
    final_mlp_kernel<<<1, 256, flds, stream>>>(g, y, F0w, F0b, F1w, F1b, L0, L1, L2w, L2b, outp);
}
